// Attention_41472204210419
// MI455X (gfx1250) — compile-verified, matrix operand order checked by reading, not run on hardware
//
#include <hip/hip_runtime.h>
#include <math.h>
#include <stdint.h>

// ---------------------------------------------------------------------------
// MI455X (gfx1250) attention block: x@wq/wk/wv (+fused RoPE+scale), flash
// attention with online softmax (row-sum via WMMA against a ones matrix) and
// async-to-LDS K/V staging, out = attn_out @ wo.
// All matmuls via v_wmma_f32_16x16x32_bf16 (wave32), fp32 accumulation.
// ---------------------------------------------------------------------------

typedef __bf16 bf16;
typedef __attribute__((ext_vector_type(4)))  __bf16 v4bf;
typedef __attribute__((ext_vector_type(8)))  __bf16 v8bf;
typedef __attribute__((ext_vector_type(16))) __bf16 v16bf;
typedef __attribute__((ext_vector_type(8)))  float  v8f;

#define DEVINL __device__ __forceinline__

constexpr int BATCH = 4;
constexpr int SEQ   = 1024;
constexpr int HID   = 4096;
constexpr int HDIM  = 128;
constexpr int HQ    = 32;
constexpr int HKV   = 8;
constexpr int MROWS = BATCH * SEQ;        // 4096
constexpr float LOG2_ROPE_BASE = 13.287712379549449f;   // log2(10000)
constexpr float ATT_SCALE = 0.08838834764831845f;       // 1/sqrt(128)

DEVINL v8f wmma_bf16(v16bf a, v16bf b, v8f c) {
  return __builtin_amdgcn_wmma_f32_16x16x32_bf16(
      false, a, false, b, (short)0, c, false, false);
}

// Load one 16x32 bf16 fragment (A layout; B mirrored with n<->m):
// lane = 16c + m; VGPRs 0..3 hold dims {8c+0..7}, VGPRs 4..7 hold {16+8c..+7}.
DEVINL v16bf load_frag(const bf16* __restrict__ base, int stride, int lane) {
  const int m = lane & 15;
  const int c = lane >> 4;
  const bf16* p = base + (size_t)m * stride + (c << 3);
  v8bf lo = *(const v8bf*)(p);
  v8bf hi = *(const v8bf*)(p + 16);
  return __builtin_shufflevector(lo, hi, 0,1,2,3,4,5,6,7,8,9,10,11,12,13,14,15);
}

// One lane moves 16 bytes global -> LDS, tracked by ASYNCcnt.
DEVINL void async_copy_b128(uint32_t lds_byte_off, const bf16* g) {
  asm volatile("global_load_async_to_lds_b128 %0, %1, off"
               :: "v"(lds_byte_off), "v"((unsigned long long)(uintptr_t)g)
               : "memory");
}

// ---------------------------------------------------------------------------
// fp32 -> bf16 straight conversion (vectorized x4)
// ---------------------------------------------------------------------------
__global__ __launch_bounds__(256) void cvt_f32_bf16(const float* __restrict__ x,
                                                    bf16* __restrict__ xb, int n4) {
  int i = blockIdx.x * blockDim.x + threadIdx.x;
  if (i < n4) {
    float4 f = ((const float4*)x)[i];
    v4bf o = { (bf16)f.x, (bf16)f.y, (bf16)f.z, (bf16)f.w };
    ((v4bf*)xb)[i] = o;
  }
}

// ---------------------------------------------------------------------------
// Tiled transpose + convert: w[K][N] fp32 -> wT[N][K] bf16  (32x32 LDS tile)
// ---------------------------------------------------------------------------
__global__ __launch_bounds__(256) void transpose_cvt(const float* __restrict__ w,
                                                     bf16* __restrict__ wT,
                                                     int K, int N) {
  __shared__ float t[32][33];
  const int tx = threadIdx.x, ty = threadIdx.y;       // (32, 8)
  const int nb = blockIdx.x * 32, kb = blockIdx.y * 32;
#pragma unroll
  for (int i = 0; i < 32; i += 8)
    t[ty + i][tx] = w[(size_t)(kb + ty + i) * N + (nb + tx)];
  __syncthreads();
#pragma unroll
  for (int i = 0; i < 32; i += 8)
    wT[(size_t)(nb + ty + i) * K + (kb + tx)] = (bf16)t[tx][ty + i];
}

// ---------------------------------------------------------------------------
// bf16 WMMA GEMM: C[4096, N] = A[4096, K] * Bt[N, K]^T
// 256 threads = 8 waves (2 M x 4 N); block tile 128x256; wave tile 64x64.
// All 8 fragment loads issued before the 16 WMMAs so VMEM stays pipelined.
// MODE: 0 = fp32 store, 1 = fused RoPE (+oscale) -> bf16 [B,H,S,D],
//       2 = V^T -> [B,HKV,D,S]
// ---------------------------------------------------------------------------
template <int MODE>
__global__ __launch_bounds__(256) void gemm_bf16(const bf16* __restrict__ A,
                                                 const bf16* __restrict__ Bt,
                                                 void* __restrict__ Cout,
                                                 const int* __restrict__ spp,
                                                 int N, int K, int H,
                                                 float oscale) {
  const int lane = threadIdx.x & 31;
  const int wave = threadIdx.x >> 5;                  // 0..7
  const int m0 = blockIdx.y * 128 + (wave >> 2) * 64;
  const int n0 = blockIdx.x * 256 + (wave & 3) * 64;

  v8f acc[4][4] = {};

  const bf16* aRow = A + (size_t)m0 * K;
  for (int k0 = 0; k0 < K; k0 += 32) {
    if (k0 + 64 <= K) __builtin_prefetch(aRow + k0 + 32, 0, 0);  // global_prefetch_b8
    v16bf a[4], bfr[4];
#pragma unroll
    for (int mt = 0; mt < 4; ++mt)
      a[mt] = load_frag(aRow + (size_t)(16 * mt) * K + k0, K, lane);
#pragma unroll
    for (int nt = 0; nt < 4; ++nt)
      bfr[nt] = load_frag(Bt + (size_t)(n0 + 16 * nt) * K + k0, K, lane);
#pragma unroll
    for (int nt = 0; nt < 4; ++nt)
#pragma unroll
      for (int mt = 0; mt < 4; ++mt)
        acc[mt][nt] = wmma_bf16(a[mt], bfr[nt], acc[mt][nt]);
  }

  // Epilogue. C layout: VGPR r, lane = 16*half + n -> (M = r + 8*half, N = n)
  const int ln = lane & 15, half = lane >> 4;
  int sp0 = 0;
  if (MODE == 1) sp0 = spp[0];
  (void)H; (void)oscale;
#pragma unroll
  for (int mt = 0; mt < 4; ++mt) {
#pragma unroll
    for (int nt = 0; nt < 4; ++nt) {
#pragma unroll
      for (int r = 0; r < 8; ++r) {
        float val = acc[mt][nt][r];
        const int row = m0 + 16 * mt + r + 8 * half;   // token index b*S+s
        const int col = n0 + 16 * nt + ln;             // feature index h*D+d
        if (MODE == 0) {
          ((float*)Cout)[(size_t)row * N + col] = val;
        } else if (MODE == 1) {
          // RoPE: pairs (2i, 2i+1) are adjacent columns -> adjacent lanes.
          float partner = __shfl_xor(val, 1, 32);
          const int s = row & (SEQ - 1), bi = row >> 10;
          const int h = col >> 7, d = col & (HDIM - 1);
          float inv = exp2f((float)(d & ~1) * (-LOG2_ROPE_BASE / (float)HDIM));
          float ang = (float)(sp0 + s) * inv;
          float sn, cs;
          sincosf(ang, &sn, &cs);
          // reference: y1 = x1*sin - x2*cos ; y2 = x1*cos + x2*sin
          float y = (d & 1) ? (partner * cs + val * sn)
                            : (val * sn - partner * cs);
          // oscale folds the attention 1/sqrt(D) into Q (rotation commutes)
          ((bf16*)Cout)[(((size_t)bi * H + h) * SEQ + s) * HDIM + d] =
              (bf16)(y * oscale);
        } else {
          const int s = row & (SEQ - 1), bi = row >> 10;
          const int h = col >> 7, d = col & (HDIM - 1);
          ((bf16*)Cout)[(((size_t)bi * HKV + h) * HDIM + d) * SEQ + s] = (bf16)val;
        }
      }
    }
  }
}

// ---------------------------------------------------------------------------
// Flash attention (no mask).  128 threads = 4 waves; the 4 waves handle four
// consecutive 16-row Q blocks of the SAME (b, head) and share K/V tiles,
// double-buffer staged into LDS with global_load_async_to_lds_b128 (ASYNCcnt).
// Per 32-key step: 8 QK^T WMMAs + 8 PV WMMAs + 1 ones-WMMA (softmax row sum).
// ---------------------------------------------------------------------------
__global__ __launch_bounds__(128) void attn_flash(const bf16* __restrict__ Qr,
                                                  const bf16* __restrict__ Kr,
                                                  const bf16* __restrict__ Vt,
                                                  bf16* __restrict__ Out) {
  __shared__ __align__(16) bf16 kbuf[2][32 * 128];   // [key][dim]  8KB each
  __shared__ __align__(16) bf16 vbuf[2][128 * 32];   // [dim][key]  8KB each
  __shared__ __align__(16) bf16 pbuf[4 * 16 * 32];   // per-wave P bounce

  const int tid  = threadIdx.x;
  const int lane = tid & 31;
  const int w    = tid >> 5;
  const int gid  = blockIdx.x * 4 + w;
  const int rb   = gid & 63;            // 16-row Q block
  const int h    = (gid >> 6) & 31;     // query head (same for all 4 waves)
  const int b    = gid >> 11;           // batch
  const int kvh  = h >> 2;              // GQA: kv head = h / 4

  const bf16* Qh = Qr + ((size_t)(b * HQ + h) * SEQ + rb * 16) * HDIM;
  const bf16* Kh = Kr + (size_t)(b * HKV + kvh) * SEQ * HDIM;
  const bf16* Vh = Vt + (size_t)(b * HKV + kvh) * HDIM * SEQ;
  bf16* pw = &pbuf[w * 512];

  const uint32_t kldsBase = (uint32_t)(uintptr_t)(&kbuf[0][0]);
  const uint32_t vldsBase = (uint32_t)(uintptr_t)(&vbuf[0][0]);

  // Cooperative async stage of one 32-key K tile (contiguous 8KB) and one
  // V^T tile (128 rows x 64B) into LDS buffer `bufi`; 8 x b128 per thread.
  auto stage = [&](int bufi, int j0) {
    const bf16* gK = Kh + (size_t)j0 * HDIM;
    uint32_t kd = kldsBase + (uint32_t)bufi * (32 * 128 * 2);
    uint32_t vd = vldsBase + (uint32_t)bufi * (128 * 32 * 2);
#pragma unroll
    for (int i = 0; i < 4; ++i) {
      int idx = tid + i * 128;                 // 0..511 (16B chunks)
      async_copy_b128(kd + idx * 16, gK + idx * 8);
    }
#pragma unroll
    for (int i = 0; i < 4; ++i) {
      int idx = tid + i * 128;                 // 0..511
      int row = idx >> 2, seg = idx & 3;       // dim row, 16B segment
      async_copy_b128(vd + row * 64 + seg * 16,
                      Vh + (size_t)row * SEQ + j0 + seg * 8);
    }
  };

  stage(0, 0);                                  // prologue: tile 0 in flight

  // Preload Q fragments (global; overlaps with async stage): 16x128.
  v16bf qa[4];
#pragma unroll
  for (int c4 = 0; c4 < 4; ++c4) qa[c4] = load_frag(Qh + c4 * 32, HDIM, lane);

  // Ones fragment: row sums of P via the matrix core.
  v16bf onesf;
#pragma unroll
  for (int i = 0; i < 16; ++i) onesf[i] = (bf16)1.0f;

  v8f o[8] = {};
  float mrow[8], lrow[8];
#pragma unroll
  for (int r = 0; r < 8; ++r) { mrow[r] = -1e30f; lrow[r] = 0.0f; }

  for (int jt = 0; jt < 32; ++jt) {
    const int bufi = jt & 1;
    // Stage next tile into the other buffer (safe: barrier at end of prev
    // iteration guarantees everyone finished reading it), then wait for the
    // current tile (async loads complete in order -> <=8 outstanding).
    if (jt + 1 < 32) {
      stage(bufi ^ 1, (jt + 1) * 32);
      asm volatile("s_wait_asynccnt 0x8" ::: "memory");
    } else {
      asm volatile("s_wait_asynccnt 0x0" ::: "memory");
    }
    __syncthreads();                            // all waves' writes visible

    const bf16* kb = &kbuf[bufi][0];
    const bf16* vb = &vbuf[bufi][0];

    // S = Q * K^T.  All 8 K fragments loaded into distinct registers first so
    // the DS loads pipeline instead of serializing on s_wait_dscnt 0.
    v16bf kf[8];
#pragma unroll
    for (int c4 = 0; c4 < 4; ++c4) {
      kf[c4]     = load_frag(kb + c4 * 32, HDIM, lane);
      kf[4 + c4] = load_frag(kb + 16 * HDIM + c4 * 32, HDIM, lane);
    }
    v8f s0 = {}, s1 = {};
#pragma unroll
    for (int c4 = 0; c4 < 4; ++c4) {
      s0 = wmma_bf16(qa[c4], kf[c4], s0);
      s1 = wmma_bf16(qa[c4], kf[4 + c4], s1);
    }

    // Online softmax: 1/sqrt(D) is pre-folded into Q.  Only the row max needs
    // a lane reduction; the row sum comes from a ones-WMMA below.
    float alpha[8];
#pragma unroll
    for (int r = 0; r < 8; ++r) {
      float v0 = s0[r], v1 = s1[r];
      float mx = fmaxf(v0, v1);
#pragma unroll
      for (int off = 1; off < 16; off <<= 1) mx = fmaxf(mx, __shfl_xor(mx, off, 32));
      float mn = fmaxf(mrow[r], mx);
      float a  = __expf(mrow[r] - mn);
      float p0 = __expf(v0 - mn);
      float p1 = __expf(v1 - mn);
      mrow[r] = mn;
      alpha[r] = a;
      const int row = r + ((lane >> 4) << 3);
      const int col = lane & 15;
      pw[row * 32 + col]      = (bf16)p0;
      pw[row * 32 + 16 + col] = (bf16)p1;
    }
    asm volatile("s_wait_dscnt 0" ::: "memory");   // cross-lane LDS visibility

#pragma unroll
    for (int nt = 0; nt < 8; ++nt)
#pragma unroll
      for (int r = 0; r < 8; ++r) o[nt][r] *= alpha[r];

    // Re-read P as an A fragment (16x32)
    const int m = lane & 15, c = lane >> 4;
    const bf16* pp = pw + m * 32 + (c << 3);
    v8bf plo = *(const v8bf*)(pp);
    v8bf phi = *(const v8bf*)(pp + 16);
    v16bf pf = __builtin_shufflevector(plo, phi,
        0,1,2,3,4,5,6,7,8,9,10,11,12,13,14,15);

    // Row sum of P via matrix core: every column of the result tile holds the
    // row sum, in exactly the (register r, lane-half) slots of lrow[].
    v8f zero = {};
    v8f psum = wmma_bf16(pf, onesf, zero);

    // O += P * V  (V^T tile in LDS: [dim][key], 32 keys per row)
    v16bf vfr[8];
#pragma unroll
    for (int nt = 0; nt < 8; ++nt)
      vfr[nt] = load_frag(vb + (nt * 16) * 32, 32, lane);
#pragma unroll
    for (int nt = 0; nt < 8; ++nt)
      o[nt] = wmma_bf16(pf, vfr[nt], o[nt]);

#pragma unroll
    for (int r = 0; r < 8; ++r) lrow[r] = lrow[r] * alpha[r] + psum[r];

    __syncthreads();          // everyone done reading bufi before it is restaged
  }

  // Normalize and store bf16 to attOut[b*S+s][h*D+d]
  float inv[8];
#pragma unroll
  for (int r = 0; r < 8; ++r) inv[r] = 1.0f / lrow[r];
  const int ln = lane & 15, half = lane >> 4;
#pragma unroll
  for (int nt = 0; nt < 8; ++nt) {
#pragma unroll
    for (int r = 0; r < 8; ++r) {
      const int row = b * SEQ + rb * 16 + r + 8 * half;
      const int col = h * HDIM + nt * 16 + ln;
      Out[(size_t)row * HID + col] = (bf16)(o[nt][r] * inv[r]);
    }
  }
}

// ---------------------------------------------------------------------------
// Orchestration
// ---------------------------------------------------------------------------
extern "C" void kernel_launch(void* const* d_in, const int* in_sizes, int n_in,
                              void* d_out, int out_size, void* d_ws, size_t ws_size,
                              hipStream_t stream) {
  (void)in_sizes; (void)n_in; (void)out_size; (void)ws_size;
  const float* x  = (const float*)d_in[0];
  const float* wq = (const float*)d_in[1];
  const float* wk = (const float*)d_in[2];
  const float* wv = (const float*)d_in[3];
  const float* wo = (const float*)d_in[4];
  const int*   sp = (const int*)d_in[5];
  float* out = (float*)d_out;

  char* ws = (char*)d_ws;
  bf16* xb   = (bf16*)(ws);                 // 33.55 MB  x bf16 [4096,4096]
  bf16* wqT  = (bf16*)(ws +  33554432);     // 33.55 MB  [4096,4096]
  bf16* wkT  = (bf16*)(ws +  67108864);     //  8.39 MB  [1024,4096]
  bf16* wvT  = (bf16*)(ws +  75497472);     //  8.39 MB  [1024,4096]
  bf16* woT  = (bf16*)(ws +  83886080);     // 33.55 MB  [4096,4096]
  bf16* Qr   = (bf16*)(ws + 117440512);     // 33.55 MB  [B,HQ,S,D]
  bf16* Kr   = (bf16*)(ws + 150994944);     //  8.39 MB  [B,HKV,S,D]
  bf16* Vt   = (bf16*)(ws + 159383552);     //  8.39 MB  [B,HKV,D,S]
  bf16* attO = xb;  // x no longer needed after the V GEMM -> reuse

  // 1) precision conversion + weight transposes
  cvt_f32_bf16<<<16384, 256, 0, stream>>>(x, xb, (MROWS * HID) / 4);
  transpose_cvt<<<dim3(128, 128), dim3(32, 8), 0, stream>>>(wq, wqT, HID, HID);
  transpose_cvt<<<dim3(32, 128),  dim3(32, 8), 0, stream>>>(wk, wkT, HID, 1024);
  transpose_cvt<<<dim3(32, 128),  dim3(32, 8), 0, stream>>>(wv, wvT, HID, 1024);
  transpose_cvt<<<dim3(128, 128), dim3(32, 8), 0, stream>>>(wo, woT, HID, HID);

  // 2) QKV projections (RoPE+scale fused for Q, RoPE for K, transpose for V)
  gemm_bf16<1><<<dim3(16, 32), 256, 0, stream>>>(xb, wqT, (void*)Qr, sp, 4096, HID, HQ, ATT_SCALE);
  gemm_bf16<1><<<dim3(4, 32),  256, 0, stream>>>(xb, wkT, (void*)Kr, sp, 1024, HID, HKV, 1.0f);
  gemm_bf16<2><<<dim3(4, 32),  256, 0, stream>>>(xb, wvT, (void*)Vt, sp, 1024, HID, HKV, 1.0f);

  // 3) flash attention: 8192 waves, 4 per block sharing K/V via async LDS
  attn_flash<<<2048, 128, 0, stream>>>(Qr, Kr, Vt, attO);

  // 4) output projection -> fp32
  gemm_bf16<0><<<dim3(16, 32), 256, 0, stream>>>(attO, woT, (void*)out, sp, 4096, HID, 0, 1.0f);
}
